// VisionEncoderMambaBlock_64716567216495
// MI455X (gfx1250) — compile-verified, matrix operand order checked by reading, not run on hardware
//
#include <hip/hip_runtime.h>
#include <hip/hip_bf16.h>

#define BATCH 4
#define SEQ   4096
#define DIM   256
#define DT_RANK 16
#define D_STATE 16
#define NTOK  1024
#define NC    32      // scan chunks
#define CL    128     // chunk length (SEQ/NC)
#define ST    32      // staged steps per LDS tile

#define BM     128    // GEMM block rows
#define APITCH 40     // LDS tile pitch in bf16 elems: 32 data + 8 pad (4 dwords)

typedef __bf16 v16bf __attribute__((ext_vector_type(16)));
typedef __bf16 v8bf  __attribute__((ext_vector_type(8)));
typedef float  v8f   __attribute__((ext_vector_type(8)));
typedef unsigned int u32x4 __attribute__((ext_vector_type(4)));
typedef int          i32x4 __attribute__((ext_vector_type(4)));
typedef int          i32x8 __attribute__((ext_vector_type(8)));

__device__ __forceinline__ float softplus_f(float v) {
    return v > 20.f ? v : log1pf(__expf(v));
}
__device__ __forceinline__ float silu_f(float v) {
    return v / (1.f + __expf(-v));
}

#if __has_builtin(__builtin_amdgcn_tensor_load_to_lds)
#define USE_TDM 1
#else
#define USE_TDM 0
#endif

#if USE_TDM
// 2D tile DMA: global (row-major, rowStride elems, bf16) -> LDS, with 4-dword
// padding appended per 16-dword (32-elem) row => LDS pitch = APITCH elems.
__device__ __forceinline__ void tdm_load_2d(unsigned lds_addr, const void* gptr,
                                            unsigned tile_d0, unsigned tile_d1,
                                            unsigned strideElems) {
    unsigned long long ga = (unsigned long long)(uintptr_t)gptr;
    u32x4 g0;
    g0[0] = 1u;                                   // count=1, no gather
    g0[1] = lds_addr;                             // LDS byte address
    g0[2] = (unsigned)(ga & 0xffffffffu);         // global_addr[31:0]
    g0[3] = (unsigned)((ga >> 32) & 0x01ffffffu) | (2u << 30); // addr[56:32] | type=2
    i32x8 g1;
    g1[0] = (int)((1u << 16) | (1u << 20) | (3u << 22) | (3u << 25));
            // data_size=2B | pad_enable | pad_interval=16dw | pad_amount=4dw
    g1[1] = (int)((tile_d0 & 0xffffu) << 16);               // tensor_dim0[15:0]
    g1[2] = (int)(((tile_d0 >> 16) & 0xffffu) | ((tile_d1 & 0xffffu) << 16));
    g1[3] = (int)(((tile_d1 >> 16) & 0xffffu) | (tile_d0 << 16));  // tile_dim0
    g1[4] = (int)tile_d1;                                    // tile_dim1 (dim2=0)
    g1[5] = (int)strideElems;                                // tensor_dim0_stride
    g1[6] = 0;
    g1[7] = 0;
    i32x4 z4 = {0, 0, 0, 0};
#if defined(__clang_major__) && (__clang_major__ >= 23)
    i32x8 z8 = {0, 0, 0, 0, 0, 0, 0, 0};
    __builtin_amdgcn_tensor_load_to_lds(g0, g1, z4, z4, z8, 0);
#else
    __builtin_amdgcn_tensor_load_to_lds(g0, g1, z4, z4, 0);
#endif
}
#endif

// ---------------- f32 -> bf16 weight conversion ----------------
__global__ __launch_bounds__(256) void cvt_bf16(const float* __restrict__ s,
                                                __bf16* __restrict__ d, int n) {
    int i = blockIdx.x * 256 + threadIdx.x;
    if (i < n) d[i] = (__bf16)s[i];
}

// ---------------- LayerNorm: x (f32) -> xn (bf16) ----------------
__global__ __launch_bounds__(256) void ln_kernel(const float* __restrict__ x,
                                                 const float* __restrict__ g,
                                                 const float* __restrict__ bta,
                                                 __bf16* __restrict__ xn) {
    int t = blockIdx.x;
    int d = threadIdx.x;
    float v = x[(size_t)t * DIM + d];
    __shared__ float s1[256], s2[256];
    s1[d] = v; s2[d] = v * v;
    __syncthreads();
    for (int off = 128; off > 0; off >>= 1) {
        if (d < off) { s1[d] += s1[d + off]; s2[d] += s2[d + off]; }
        __syncthreads();
    }
    float mu  = s1[0] * (1.f / DIM);
    float var = s2[0] * (1.f / DIM) - mu * mu;
    float inv = rsqrtf(var + 1e-5f);
    xn[(size_t)t * DIM + d] = (__bf16)((v - mu) * inv * g[d] + bta[d]);
}

// ---------------- Gaussian mask (1024) ----------------
__global__ void mask_kernel(float* __restrict__ mask) {
    int t = threadIdx.x;
    const float c = 512.f, sig = 256.f;   // mean|i-512| over 0..1023 == 256 exactly
    float dv = (float)t - c;
    float w = __expf(-0.5f * dv * dv / (sig * sig));
    __shared__ float s[1024];
    s[t] = w;
    __syncthreads();
    for (int off = 512; off > 0; off >>= 1) {
        if (t < off) s[t] += s[t + off];
        __syncthreads();
    }
    mask[t] = w / s[0];
}

// ---------------- bf16 WMMA GEMM with TDM-staged LDS tiles ----------------
// out[m,n] = act(A[m,:]·W[n,:] + bias[n]).  A:(M,K) bf16, W:(N,K) bf16, both
// row-major.  Block: 8 waves, tile 128(M) x NT*16(N); wave w owns M-tile w and
// all NT N-tiles (A fragment reused NT times).  K staged in 32-wide slices,
// double-buffered in LDS via tensor_load_to_lds (TENSORcnt) when available.
template <int NT>
__global__ __launch_bounds__(256) void gemm_wmma(const __bf16* __restrict__ A,
                                                 const __bf16* __restrict__ Wb,
                                                 const float*  __restrict__ bias,
                                                 float*  __restrict__ outF,
                                                 __bf16* __restrict__ outB,
                                                 int N, int K, int act) {
    __shared__ __bf16 ldsA[2][BM * APITCH];
    __shared__ __bf16 ldsB[2][64 * APITCH];
    const int wave = threadIdx.x >> 5;
    const int lane = threadIdx.x & 31;
    const int hf   = lane >> 4;
    const int l16  = lane & 15;
    const int m0 = blockIdx.x * BM;
    const int n0 = blockIdx.y * NT * 16;
    const int nrows = NT * 16;

    const __bf16* aTile = A  + (size_t)m0 * K;
    const __bf16* bTile = Wb + (size_t)n0 * K;
    const int nIter = K / 32;

    v8f acc[NT] = {};

#if USE_TDM
    unsigned ldsAad[2], ldsBad[2];
    ldsAad[0] = (unsigned)(uintptr_t)(&ldsA[0][0]);
    ldsAad[1] = (unsigned)(uintptr_t)(&ldsA[1][0]);
    ldsBad[0] = (unsigned)(uintptr_t)(&ldsB[0][0]);
    ldsBad[1] = (unsigned)(uintptr_t)(&ldsB[1][0]);
    if (wave == 0) {
        tdm_load_2d(ldsAad[0], aTile, 32, BM,    (unsigned)K);
        tdm_load_2d(ldsBad[0], bTile, 32, nrows, (unsigned)K);
    }
#endif

    for (int it = 0; it < nIter; ++it) {
        const int cur = it & 1;
#if USE_TDM
        if (wave == 0) {
            if (it + 1 < nIter) {
                tdm_load_2d(ldsAad[cur ^ 1], aTile + (it + 1) * 32, 32, BM, (unsigned)K);
                tdm_load_2d(ldsBad[cur ^ 1], bTile + (it + 1) * 32, 32, nrows, (unsigned)K);
                __builtin_amdgcn_s_wait_tensorcnt((short)2);
            } else {
                __builtin_amdgcn_s_wait_tensorcnt((short)0);
            }
        }
        __syncthreads();
#else
        {
            const __bf16* aS = aTile + it * 32;
            for (int idx = threadIdx.x; idx < BM * 32; idx += 256) {
                int r = idx >> 5, kk = idx & 31;
                ldsA[cur][r * APITCH + kk] = aS[(size_t)r * K + kk];
            }
            const __bf16* bS = bTile + it * 32;
            for (int idx = threadIdx.x; idx < nrows * 32; idx += 256) {
                int r = idx >> 5, kk = idx & 31;
                ldsB[cur][r * APITCH + kk] = bS[(size_t)r * K + kk];
            }
        }
        __syncthreads();
#endif
        // A fragment: row = wave*16 + l16; K-halves per ISA A layout.
        const __bf16* ar = &ldsA[cur][(wave * 16 + l16) * APITCH];
        v8bf a0 = *(const v8bf*)(ar + hf * 8);
        v8bf a1 = *(const v8bf*)(ar + 16 + hf * 8);
        v16bf aF;
#pragma unroll
        for (int i = 0; i < 8; i++) { aF[i] = a0[i]; aF[i + 8] = a1[i]; }
#pragma unroll
        for (int nt = 0; nt < NT; ++nt) {
            const __bf16* br = &ldsB[cur][(nt * 16 + l16) * APITCH];
            v8bf b0 = *(const v8bf*)(br + hf * 8);
            v8bf b1 = *(const v8bf*)(br + 16 + hf * 8);
            v16bf bF;
#pragma unroll
            for (int i = 0; i < 8; i++) { bF[i] = b0[i]; bF[i + 8] = b1[i]; }
            acc[nt] = __builtin_amdgcn_wmma_f32_16x16x32_bf16(false, aF, false, bF,
                                                              (short)0, acc[nt],
                                                              false, false);
        }
        __syncthreads();
    }

#pragma unroll
    for (int nt = 0; nt < NT; ++nt) {
        int n = n0 + nt * 16 + l16;
        float bv = bias ? bias[n] : 0.f;
#pragma unroll
        for (int r = 0; r < 8; r++) {
            int m = m0 + wave * 16 + hf * 8 + r;
            float v = acc[nt][r] + bv;
            if (act == 1) v = softplus_f(v);
            size_t o = (size_t)m * N + n;
            if (outF) outF[o] = v;
            if (outB) outB[o] = (__bf16)v;
        }
    }
}

// ---------------- pool z (4:1 over seq) + SiLU ----------------
__global__ __launch_bounds__(256) void zpool_kernel(const float* __restrict__ z,
                                                    float* __restrict__ zp) {
    int idx = blockIdx.x * 256 + threadIdx.x;       // B*T*D
    int d = idx & 255, t = (idx >> 8) & 1023, b = idx >> 18;
    size_t base = ((size_t)(b * SEQ + t * 4)) * DIM + d;
    float a = 0.f;
#pragma unroll
    for (int j = 0; j < 4; j++) a += z[base + (size_t)j * DIM];
    a *= 0.25f;
    zp[idx] = silu_f(a);
}

// ---------------- chunked selective scan (pass1 / pass3) ----------------
// dir=0: left->right scan of h1.  dir=1: right->left scan of h2 (equivalent to
// the reference's reverse/scan/reverse, all in original orientation).
// grid: (NC, BATCH, 2); 256 threads = one per channel d.
__global__ __launch_bounds__(256) void scan_kernel(
    const float* __restrict__ dbc_f, const float* __restrict__ dbc_b,
    const __bf16* __restrict__ h1b,  const __bf16* __restrict__ h2b,
    const float* __restrict__ Wdt_f, const float* __restrict__ bdt_f,
    const float* __restrict__ Alog_f, const float* __restrict__ Dv_f,
    const float* __restrict__ Wdt_b, const float* __restrict__ bdt_b,
    const float* __restrict__ Alog_b, const float* __restrict__ Dv_b,
    float* __restrict__ cs, float* __restrict__ y1p, float* __restrict__ y2p,
    int pass3) {
    int c = blockIdx.x, b = blockIdx.y, dir = blockIdx.z;
    int d = threadIdx.x;
    const float*  dbc  = dir ? dbc_b : dbc_f;
    const __bf16* hx   = dir ? h2b   : h1b;
    const float*  Wdt  = dir ? Wdt_b : Wdt_f;
    const float*  bdt  = dir ? bdt_b : bdt_f;
    const float*  Alog = dir ? Alog_b : Alog_f;
    const float*  Dv   = dir ? Dv_b  : Dv_f;
    float*        yp   = dir ? y2p   : y1p;

    float wdt[DT_RANK], arr[D_STATE], h[D_STATE], ap[D_STATE];
#pragma unroll
    for (int r = 0; r < DT_RANK; r++) wdt[r] = Wdt[d * DT_RANK + r];
#pragma unroll
    for (int n = 0; n < D_STATE; n++) arr[n] = -__expf(Alog[d * D_STATE + n]);
    float bd = bdt[d], Dd = Dv[d];

    size_t csBase = ((size_t)((dir * BATCH + b) * NC + c)) * (DIM * D_STATE * 2)
                  + (size_t)d * (D_STATE * 2);
    if (pass3) {
#pragma unroll
        for (int n = 0; n < D_STATE; n++) h[n] = cs[csBase + 2 * n + 1];
    } else {
#pragma unroll
        for (int n = 0; n < D_STATE; n++) { h[n] = 0.f; ap[n] = 1.f; }
    }

    __shared__ float ld[ST * 48];
    const int s0 = c * CL;
    const int nTiles = CL / ST;
    float acc4 = 0.f;
    for (int tt = 0; tt < nTiles; ++tt) {
        int sub = dir ? (CL - ST - tt * ST) : (tt * ST);
        __syncthreads();
        size_t gb = ((size_t)(b * SEQ + s0 + sub)) * 48;
        for (int i = threadIdx.x; i < ST * 48; i += 256) ld[i] = dbc[gb + i];
        if (tt + 1 < nTiles) {   // prefetch next dbc tile (global_prefetch_b8)
            int sub2 = dir ? (sub - ST) : (sub + ST);
            size_t gb2 = ((size_t)(b * SEQ + s0 + sub2)) * 48;
            if (threadIdx.x < 48)
                __builtin_prefetch(&dbc[gb2 + (size_t)threadIdx.x * 32], 0, 0);
        }
        __syncthreads();
        for (int q = 0; q < ST; ++q) {
            int j = dir ? (ST - 1 - q) : q;
            int s = s0 + sub + j;
            const float* row = &ld[j * 48];
            float dp = bd;
#pragma unroll
            for (int r = 0; r < DT_RANK; r++) dp = fmaf(row[r], wdt[r], dp);
            float delta = softplus_f(dp);
            float xv = (float)hx[((size_t)(b * SEQ + s)) * DIM + d];
            float y = 0.f;
#pragma unroll
            for (int n = 0; n < D_STATE; n++) {
                float dA = __expf(delta * arr[n]);
                h[n] = dA * h[n] + (delta * row[16 + n]) * xv;
                if (!pass3) ap[n] *= dA;
                else        y = fmaf(h[n], row[32 + n], y);
            }
            if (pass3) {
                acc4 += y + Dd * xv;
                bool wr = dir ? ((j & 3) == 0) : ((j & 3) == 3);
                if (wr) {
                    yp[((size_t)(b * NTOK + (s >> 2))) * DIM + d] = acc4 * 0.25f;
                    acc4 = 0.f;
                }
            }
        }
    }
    if (!pass3) {
#pragma unroll
        for (int n = 0; n < D_STATE; n++) {
            cs[csBase + 2 * n]     = ap[n];
            cs[csBase + 2 * n + 1] = h[n];
        }
    }
}

// ---------------- pass2: per-chain prefix over chunk states ----------------
__global__ __launch_bounds__(256) void scan_fix(float* __restrict__ cs) {
    int idx = blockIdx.x * 256 + threadIdx.x;      // 2*B*D*N = 32768
    int n = idx & 15, d = (idx >> 4) & 255, bd = idx >> 12;  // bd = dir*4+b
    int dirf = bd >> 2;                            // chunk order follows scan dir
    float hin = 0.f;
    for (int i = 0; i < NC; i++) {
        int c = dirf ? (NC - 1 - i) : i;
        size_t o = ((size_t)(bd * NC + c)) * (DIM * D_STATE * 2)
                 + (size_t)d * (D_STATE * 2) + 2 * n;
        float a = cs[o], hp = cs[o + 1];
        cs[o + 1] = hin;                           // h-slot becomes chunk init state
        hin = a * hin + hp;
    }
}

// ---------------- final combine ----------------
__global__ __launch_bounds__(256) void final_kernel(const float* __restrict__ x,
                                                    const float* __restrict__ zp,
                                                    const float* __restrict__ y1p,
                                                    const float* __restrict__ y2p,
                                                    const float* __restrict__ mask,
                                                    float* __restrict__ out) {
    int idx = blockIdx.x * 256 + threadIdx.x;      // B*T*D
    int d = idx & 255, t = (idx >> 8) & 1023, b = idx >> 18;
    size_t xb = ((size_t)(b * SEQ + t * 4)) * DIM + d;
    float sk = 0.f;
#pragma unroll
    for (int j = 0; j < 4; j++) sk += x[xb + (size_t)j * DIM];
    sk *= 0.25f;
    float m = mask[t];
    float a1 = y1p[idx] * m;
    float a2 = y2p[idx] * m;                       // backward path already in
    out[idx] = zp[idx] * (silu_f(a1) + silu_f(a2)) + sk;   // original orientation
}

extern "C" void kernel_launch(void* const* d_in, const int* in_sizes, int n_in,
                              void* d_out, int out_size, void* d_ws, size_t ws_size,
                              hipStream_t stream) {
    const float* x      = (const float*)d_in[0];
    const float* ln_g   = (const float*)d_in[1];
    const float* ln_b   = (const float*)d_in[2];
    const float* W1     = (const float*)d_in[3];
    const float* b1     = (const float*)d_in[4];
    const float* W2     = (const float*)d_in[5];
    const float* b2     = (const float*)d_in[6];
    const float* Wcf    = (const float*)d_in[7];
    const float* bcf    = (const float*)d_in[8];
    const float* Wcb    = (const float*)d_in[9];
    const float* bcb    = (const float*)d_in[10];
    const float* Wdbc_f = (const float*)d_in[11];
    const float* Wdt_f  = (const float*)d_in[12];
    const float* bdt_f  = (const float*)d_in[13];
    const float* Alog_f = (const float*)d_in[14];
    const float* D_f    = (const float*)d_in[15];
    const float* Wdbc_b = (const float*)d_in[16];
    const float* Wdt_b  = (const float*)d_in[17];
    const float* bdt_b  = (const float*)d_in[18];
    const float* Alog_b = (const float*)d_in[19];
    const float* D_b    = (const float*)d_in[20];
    float* out = (float*)d_out;

    const size_t NF = (size_t)BATCH * SEQ * DIM;          // 4,194,304
    char* ws = (char*)d_ws;
    size_t off = 0;
    __bf16* xn_bf = (__bf16*)(ws + off); off += NF * 2;
    __bf16* xp_bf = (__bf16*)(ws + off); off += NF * 2;
    __bf16* h1_bf = (__bf16*)(ws + off); off += NF * 2;
    __bf16* h2_bf = (__bf16*)(ws + off); off += NF * 2;
    float*  z     = (float*)(ws + off);  off += NF * 4;
    float*  dbc_f = (float*)(ws + off);  off += (size_t)BATCH * SEQ * 48 * 4;
    float*  dbc_b = (float*)(ws + off);  off += (size_t)BATCH * SEQ * 48 * 4;
    float*  cs    = (float*)(ws + off);  off += (size_t)2 * BATCH * NC * DIM * D_STATE * 2 * 4;
    float*  y1p   = (float*)(ws + off);  off += (size_t)BATCH * NTOK * DIM * 4;
    float*  y2p   = (float*)(ws + off);  off += (size_t)BATCH * NTOK * DIM * 4;
    float*  zp    = (float*)(ws + off);  off += (size_t)BATCH * NTOK * DIM * 4;
    float*  mask  = (float*)(ws + off);  off += NTOK * 4;
    __bf16* W1bf  = (__bf16*)(ws + off); off += (size_t)DIM * DIM * 2;
    __bf16* W2bf  = (__bf16*)(ws + off); off += (size_t)DIM * DIM * 2;
    __bf16* Wcfbf = (__bf16*)(ws + off); off += (size_t)DIM * DIM * 2;
    __bf16* Wcbbf = (__bf16*)(ws + off); off += (size_t)DIM * DIM * 2;
    __bf16* Wdfbf = (__bf16*)(ws + off); off += (size_t)48 * DIM * 2;
    __bf16* Wdbbf = (__bf16*)(ws + off); off += (size_t)48 * DIM * 2;
    (void)ws_size; (void)in_sizes; (void)n_in; (void)out_size;

    // 0) weight conversion f32 -> bf16 (once per launch)
    cvt_bf16<<<DIM * DIM / 256, 256, 0, stream>>>(W1,  W1bf,  DIM * DIM);
    cvt_bf16<<<DIM * DIM / 256, 256, 0, stream>>>(W2,  W2bf,  DIM * DIM);
    cvt_bf16<<<DIM * DIM / 256, 256, 0, stream>>>(Wcf, Wcfbf, DIM * DIM);
    cvt_bf16<<<DIM * DIM / 256, 256, 0, stream>>>(Wcb, Wcbbf, DIM * DIM);
    cvt_bf16<<<48 * DIM / 256, 256, 0, stream>>>(Wdbc_f, Wdfbf, 48 * DIM);
    cvt_bf16<<<48 * DIM / 256, 256, 0, stream>>>(Wdbc_b, Wdbbf, 48 * DIM);

    // 1) layernorm -> xn (bf16), gaussian mask
    ln_kernel<<<BATCH * SEQ, 256, 0, stream>>>(x, ln_g, ln_b, xn_bf);
    mask_kernel<<<1, 1024, 0, stream>>>(mask);

    // 2) WMMA GEMMs (TDM double-buffered LDS tiles)
    dim3 gFull(BATCH * SEQ / BM, DIM / 64);   // NT=4
    dim3 gDbc (BATCH * SEQ / BM, 1);          // NT=3 (N=48)
    gemm_wmma<4><<<gFull, 256, 0, stream>>>(xn_bf, W1bf,  b1,  z,      nullptr, DIM, DIM, 0);
    gemm_wmma<4><<<gFull, 256, 0, stream>>>(xn_bf, W2bf,  b2,  nullptr, xp_bf,  DIM, DIM, 0);
    gemm_wmma<4><<<gFull, 256, 0, stream>>>(xp_bf, Wcfbf, bcf, nullptr, h1_bf,  DIM, DIM, 1);
    gemm_wmma<4><<<gFull, 256, 0, stream>>>(xp_bf, Wcbbf, bcb, nullptr, h2_bf,  DIM, DIM, 1);
    gemm_wmma<3><<<gDbc, 256, 0, stream>>>(h1_bf, Wdfbf, nullptr, dbc_f, nullptr, 48, DIM, 0);
    gemm_wmma<3><<<gDbc, 256, 0, stream>>>(h2_bf, Wdbbf, nullptr, dbc_b, nullptr, 48, DIM, 0);

    // 3) pooled z path
    zpool_kernel<<<(BATCH * NTOK * DIM) / 256, 256, 0, stream>>>(z, zp);

    // 4) chunked selective scan: pass1 -> pass2 (prefix) -> pass3 (y + pool)
    dim3 gScan(NC, BATCH, 2);
    scan_kernel<<<gScan, 256, 0, stream>>>(dbc_f, dbc_b, h1_bf, h2_bf,
                                           Wdt_f, bdt_f, Alog_f, D_f,
                                           Wdt_b, bdt_b, Alog_b, D_b,
                                           cs, y1p, y2p, 0);
    scan_fix<<<(2 * BATCH * DIM * D_STATE) / 256, 256, 0, stream>>>(cs);
    scan_kernel<<<gScan, 256, 0, stream>>>(dbc_f, dbc_b, h1_bf, h2_bf,
                                           Wdt_f, bdt_f, Alog_f, D_f,
                                           Wdt_b, bdt_b, Alog_b, D_b,
                                           cs, y1p, y2p, 1);

    // 5) combine: out = zp*(silu(y1p*m)+silu(y2p*m)) + skip
    final_kernel<<<(BATCH * NTOK * DIM) / 256, 256, 0, stream>>>(x, zp, y1p, y2p, mask, out);
}
